// VanillaMHA_2851858285112
// MI455X (gfx1250) — compile-verified, matrix operand order checked by reading, not run on hardware
//
#include <hip/hip_runtime.h>
#include <hip/hip_bf16.h>

// ---------------------------------------------------------------------------
// Problem constants
// ---------------------------------------------------------------------------
#define BB   4
#define SS   2048
#define DD   1024
#define HH   16
#define DHH  64
#define OUTD 1024
#define BS   (BB * SS)   // 8192 rows

typedef __attribute__((ext_vector_type(16))) __bf16         v16bf;
typedef __attribute__((ext_vector_type(16))) unsigned short v16us;
typedef __attribute__((ext_vector_type(8)))  float          v8f;
typedef __attribute__((ext_vector_type(4)))  unsigned int   v4u;
typedef __attribute__((ext_vector_type(8)))  int            v8i_t;
typedef __attribute__((ext_vector_type(4)))  int            v4i_t;

// TDM-staged weight tile in LDS: 64 rows x 32 bf16, padded to 80B/row by the
// TDM pad feature (pad_interval = 16 dwords, pad_amount = 4 dwords).
#define LDS_ROW_ELEMS 40                 // 80 bytes
#define LDS_TILE_ELEMS (64 * LDS_ROW_ELEMS)

// ---------------------------------------------------------------------------
// Helpers
// ---------------------------------------------------------------------------
__device__ __forceinline__ unsigned short f2bf(float f) {
    unsigned int u = __float_as_uint(f);
    u += 0x7FFFu + ((u >> 16) & 1u);      // round-to-nearest-even
    return (unsigned short)(u >> 16);
}

__device__ __forceinline__ v8f v8f_zero() {
    v8f z;
#pragma unroll
    for (int i = 0; i < 8; ++i) z[i] = 0.0f;
    return z;
}

union FragU {
    v16bf bf;
    v16us us;
    uint4 q[2];
};

// 16-element bf16 fragment: elements [0..7] = p[0..7], [8..15] = p[16..23].
// Matches CDNA5 16-bit A/B operand layout when p is pre-offset by 8*(lane>>4).
__device__ __forceinline__ v16bf load_frag(const unsigned short* p) {
    FragU f;
    f.q[0] = *(const uint4*)(p);
    f.q[1] = *(const uint4*)(p + 16);
    return f.bf;
}

__device__ __forceinline__ v8f wmma_bf16(v16bf a, v16bf b, v8f c) {
    return __builtin_amdgcn_wmma_f32_16x16x32_bf16(
        false, a, false, b, (short)0, c, false, false);
}

// ---------------------------------------------------------------------------
// TDM: DMA one 64(N) x 32(K) bf16 weight tile into LDS (padded rows).
// D# per CDNA5 ISA ch.8: group0 = {count,lds_addr,global_addr,type=2},
// group1 = {data_size=2B + pad cfg, tensor dims, tile dims, dim0 stride}.
// Groups 2/3 unused for 2-D tiles (tile_dim2 = 0) -> zeros.
// ---------------------------------------------------------------------------
__device__ __forceinline__ void tdm_load_tile(unsigned lds_byte_off,
                                              const unsigned short* gptr,
                                              int Ktot, int Ntot, int ldw) {
    unsigned long long ga = (unsigned long long)(size_t)gptr;
    v4u g0;
    g0[0] = 1u;                                              // count = 1
    g0[1] = lds_byte_off;                                    // lds_addr
    g0[2] = (unsigned)(ga & 0xFFFFFFFFull);                  // global_addr lo
    g0[3] = (unsigned)((ga >> 32) & 0x1FFFFFFull) | 0x80000000u; // hi | type=2
    v8i_t g1;
    // data_size=1(2B), pad_enable, pad_interval=3(16 dw), pad_amount=3(4 dw)
    g1[0] = (int)((1u << 16) | (1u << 20) | (3u << 22) | (3u << 25));
    g1[1] = (int)(((unsigned)Ktot & 0xFFFFu) << 16);         // tensor_dim0 lo
    g1[2] = (int)(((unsigned)Ktot >> 16) | (((unsigned)Ntot & 0xFFFFu) << 16));
    g1[3] = (int)(((unsigned)Ntot >> 16) | (32u << 16));     // tile_dim0 = 32
    g1[4] = 64;                                              // tile_dim1 = 64
    g1[5] = ldw;                                             // dim0 stride lo
    g1[6] = 0;
    g1[7] = 0;
    v4i_t z4 = {0, 0, 0, 0};
    v8i_t z8;
#pragma unroll
    for (int i = 0; i < 8; ++i) z8[i] = 0;
    __builtin_amdgcn_tensor_load_to_lds(g0, g1, z4, z4, z8, 0);
}

// ---------------------------------------------------------------------------
// One wave computes a 32(M) x 64(N) tile of  out = A * W^T.
// A: [M x K] bf16 row-major (global).  W: [N x K] bf16 row-major, staged
// through LDS by wave 0 via double-buffered TDM transfers.
// ---------------------------------------------------------------------------
__device__ __forceinline__ void gemm32x64_tdm(const unsigned short* A, int lda,
                                              const unsigned short* W, int ldw,
                                              int Ntot, int m0, int n0, int K,
                                              unsigned short* sm, v8f acc[2][4]) {
    const int tid  = threadIdx.x;
    const int w    = tid >> 5;
    const int lane = tid & 31;
    const int lr   = lane & 15;
    const int hi   = lane >> 4;

    const unsigned short* ap0 = A + (size_t)(m0 + lr) * lda + 8 * hi;
    const unsigned short* ap1 = ap0 + (size_t)16 * lda;
    const unsigned smbase = (unsigned)(size_t)sm;    // LDS byte offset

    if (w == 0) {
        tdm_load_tile(smbase, W + (size_t)n0 * ldw, K, Ntot, ldw);
        __builtin_amdgcn_s_wait_tensorcnt(0);
    }
    __syncthreads();

    for (int k = 0; k < K; k += 32) {
        const int cur = (k >> 5) & 1;
        if (w == 0 && (k + 32) < K) {
            tdm_load_tile(smbase + (unsigned)((cur ^ 1) * (LDS_TILE_ELEMS * 2)),
                          W + (size_t)n0 * ldw + (k + 32), K, Ntot, ldw);
        }
        const unsigned short* bsm = sm + (size_t)cur * LDS_TILE_ELEMS;
        v16bf af0 = load_frag(ap0 + k);
        v16bf af1 = load_frag(ap1 + k);
#pragma unroll
        for (int t = 0; t < 4; ++t) {
            v16bf bf = load_frag(bsm + (16 * t + lr) * LDS_ROW_ELEMS + 8 * hi);
            acc[0][t] = wmma_bf16(af0, bf, acc[0][t]);
            acc[1][t] = wmma_bf16(af1, bf, acc[1][t]);
        }
        if (w == 0 && (k + 32) < K) __builtin_amdgcn_s_wait_tensorcnt(0);
        __syncthreads();
    }
}

// ---------------------------------------------------------------------------
// Kernel 0: f32 -> bf16 elementwise convert (weights)
// ---------------------------------------------------------------------------
__global__ void __launch_bounds__(256)
cvt_bf16_kernel(const float* __restrict__ in, unsigned short* __restrict__ out, int n) {
    int i = blockIdx.x * 256 + threadIdx.x;
    if (i < n) out[i] = f2bf(in[i]);
}

// ---------------------------------------------------------------------------
// Kernel 1: LayerNorm + pos_emb; writes xp (f32 residual) and xp_bf16 (GEMM A)
// ---------------------------------------------------------------------------
__global__ void __launch_bounds__(256)
ln_pos_kernel(const float* __restrict__ x, const float* __restrict__ pos,
              const float* __restrict__ gamma, const float* __restrict__ beta,
              float* __restrict__ xp, unsigned short* __restrict__ xpb) {
    const int row = blockIdx.x;
    const int s   = row & (SS - 1);
    const int tid = threadIdx.x;
    const float* xr = x + (size_t)row * DD;

    __shared__ float red[256];

    float vals[4];
    float lsum = 0.0f;
#pragma unroll
    for (int i = 0; i < 4; ++i) {
        float v = xr[tid + 256 * i];
        vals[i] = v;
        lsum += v;
    }
    red[tid] = lsum;
    __syncthreads();
    for (int st = 128; st > 0; st >>= 1) {
        if (tid < st) red[tid] += red[tid + st];
        __syncthreads();
    }
    const float mu = red[0] * (1.0f / DD);
    __syncthreads();

    float lsq = 0.0f;
#pragma unroll
    for (int i = 0; i < 4; ++i) {
        float d = vals[i] - mu;
        lsq += d * d;
    }
    red[tid] = lsq;
    __syncthreads();
    for (int st = 128; st > 0; st >>= 1) {
        if (tid < st) red[tid] += red[tid + st];
        __syncthreads();
    }
    const float rstd = rsqrtf(red[0] * (1.0f / DD) + 1e-5f);

#pragma unroll
    for (int i = 0; i < 4; ++i) {
        int d = tid + 256 * i;
        float xn = (vals[i] - mu) * rstd * gamma[d] + beta[d];
        float v  = xn + pos[(size_t)s * DD + d];
        xp [(size_t)row * DD + d] = v;
        xpb[(size_t)row * DD + d] = f2bf(v);
    }
}

// ---------------------------------------------------------------------------
// Kernel 2: QKV GEMM -> scatter q/k/v^T per head
//   q,k : [B*H, S, DH] bf16      vT : [B*H, DH, S] bf16
// ---------------------------------------------------------------------------
__global__ void __launch_bounds__(256)
qkv_gemm_kernel(const unsigned short* __restrict__ xpb,
                const unsigned short* __restrict__ W,   // [3D x D] bf16
                const float* __restrict__ bias,
                unsigned short* __restrict__ qb,
                unsigned short* __restrict__ kb,
                unsigned short* __restrict__ vT) {
    __shared__ unsigned short smem[2 * LDS_TILE_ELEMS];
    const int w  = threadIdx.x >> 5;
    const int m0 = blockIdx.y * 256 + w * 32;
    const int n0 = blockIdx.x * 64;

    v8f acc[2][4];
#pragma unroll
    for (int s2 = 0; s2 < 2; ++s2)
#pragma unroll
        for (int t = 0; t < 4; ++t) acc[s2][t] = v8f_zero();

    gemm32x64_tdm(xpb, DD, W, DD, 3 * DD, m0, n0, DD, smem, acc);

    const int lane = threadIdx.x & 31;
    const int lr = lane & 15, hi = lane >> 4;
#pragma unroll
    for (int t = 0; t < 4; ++t) {
        const int n    = n0 + 16 * t + lr;
        const int sect = n >> 10;            // 0=q 1=k 2=v
        const int col  = n & (DD - 1);
        const int h    = col >> 6;
        const int d    = col & (DHH - 1);
        const float bn = bias[n];
#pragma unroll
        for (int s2 = 0; s2 < 2; ++s2)
#pragma unroll
            for (int r = 0; r < 8; ++r) {
                const int m = m0 + 16 * s2 + r + 8 * hi;
                const int b = m >> 11;
                const int s = m & (SS - 1);
                const unsigned short bv = f2bf(acc[s2][t][r] + bn);
                const size_t bh = (size_t)(b * HH + h);
                if (sect == 0)      qb[(bh * SS + s) * DHH + d] = bv;
                else if (sect == 1) kb[(bh * SS + s) * DHH + d] = bv;
                else                vT[(bh * DHH + d) * SS + s] = bv;
            }
    }
}

// ---------------------------------------------------------------------------
// Kernel 3: attention.  One wave = 16 queries of one (b,h), online softmax.
// Computes S^T = K * Q^T so the score C-layout IS the A-layout needed for P
// in the P*V WMMA (zero data movement between the two matmuls).
// ---------------------------------------------------------------------------
__global__ void __launch_bounds__(256)
attn_kernel(const unsigned short* __restrict__ qb,
            const unsigned short* __restrict__ kb,
            const unsigned short* __restrict__ vT,
            const float* __restrict__ mask,          // [B,S], 1 = valid
            unsigned short* __restrict__ o) {
    const int w    = threadIdx.x >> 5;
    const int lane = threadIdx.x & 31;
    const int lr   = lane & 15;
    const int hi   = lane >> 4;

    const int gw = blockIdx.x * 8 + w;
    const int qt = gw & 127;                 // S/16 = 128 query tiles
    const int bh = gw >> 7;
    const int b  = bh >> 4;
    const int h  = bh & 15;

    const unsigned short* Q = qb + (size_t)bh * SS * DHH;
    const unsigned short* K = kb + (size_t)bh * SS * DHH;
    const unsigned short* V = vT + (size_t)bh * DHH * SS;
    const float* mrow = mask + (size_t)b * SS;
    const int q0 = qt * 16;

    v16bf qf0, qf1;
    {
        const unsigned short* qp = Q + (size_t)(q0 + lr) * DHH + 8 * hi;
        qf0 = load_frag(qp);
        qf1 = load_frag(qp + 32);
    }

    v8f o_acc[4] = {v8f_zero(), v8f_zero(), v8f_zero(), v8f_zero()};
    float m_q = -3.0e38f;
    float l_q = 0.0f;

    for (int kb0 = 0; kb0 < SS; kb0 += 32) {
        v8f st[2];
#pragma unroll
        for (int t = 0; t < 2; ++t) {
            const unsigned short* kp = K + (size_t)(kb0 + 16 * t + lr) * DHH + 8 * hi;
            v16bf a0 = load_frag(kp);
            v16bf a1 = load_frag(kp + 32);
            v8f acc = v8f_zero();
            acc = wmma_bf16(a0, qf0, acc);
            acc = wmma_bf16(a1, qf1, acc);
            st[t] = acc;
        }
        float sc[16];
        float bm = -3.0e38f;
#pragma unroll
        for (int t = 0; t < 2; ++t)
#pragma unroll
            for (int r = 0; r < 8; ++r) {
                const int key = kb0 + 16 * t + r + 8 * hi;
                float v = st[t][r] * 0.125f + (1.0f - mrow[key]);
                sc[t * 8 + r] = v;
                bm = fmaxf(bm, v);
            }
        bm = fmaxf(bm, __shfl_xor(bm, 16, 32));
        const float nm    = fmaxf(m_q, bm);
        const float alpha = __expf(m_q - nm);
        m_q = nm;

        float psum = 0.0f;
        v16us pv;
#pragma unroll
        for (int i = 0; i < 16; ++i) {
            float p = __expf(sc[i] - nm);
            psum += p;
            pv[i] = f2bf(p);
        }
        psum += __shfl_xor(psum, 16, 32);
        l_q = l_q * alpha + psum;

        float ar[8];
#pragma unroll
        for (int r = 0; r < 8; ++r) ar[r] = __shfl(alpha, 8 * hi + r, 32);
#pragma unroll
        for (int t = 0; t < 4; ++t)
#pragma unroll
            for (int r = 0; r < 8; ++r) o_acc[t][r] *= ar[r];

        const v16bf pf = __builtin_bit_cast(v16bf, pv);
#pragma unroll
        for (int t = 0; t < 4; ++t) {
            const unsigned short* vp = V + (size_t)(16 * t + lr) * SS + kb0 + 8 * hi;
            v16bf vf = load_frag(vp);
            o_acc[t] = wmma_bf16(pf, vf, o_acc[t]);
        }
    }

    float lrcp[8];
#pragma unroll
    for (int r = 0; r < 8; ++r) {
        float lv = __shfl(l_q, 8 * hi + r, 32);
        lrcp[r] = 1.0f / lv;
    }
#pragma unroll
    for (int t = 0; t < 4; ++t)
#pragma unroll
        for (int r = 0; r < 8; ++r) {
            const int m   = b * SS + q0 + r + 8 * hi;
            const int col = h * DHH + 16 * t + lr;
            o[(size_t)m * DD + col] = f2bf(o_acc[t][r] * lrcp[r]);
        }
}

// ---------------------------------------------------------------------------
// Kernel 4: out_proj GEMM + bias + residual(xp) -> y bf16 [BS x D]
// ---------------------------------------------------------------------------
__global__ void __launch_bounds__(256)
outproj_gemm_kernel(const unsigned short* __restrict__ ob,
                    const unsigned short* __restrict__ W,
                    const float* __restrict__ bias,
                    const float* __restrict__ xp,
                    unsigned short* __restrict__ y) {
    __shared__ unsigned short smem[2 * LDS_TILE_ELEMS];
    const int w  = threadIdx.x >> 5;
    const int m0 = blockIdx.y * 256 + w * 32;
    const int n0 = blockIdx.x * 64;

    v8f acc[2][4];
#pragma unroll
    for (int s2 = 0; s2 < 2; ++s2)
#pragma unroll
        for (int t = 0; t < 4; ++t) acc[s2][t] = v8f_zero();

    gemm32x64_tdm(ob, DD, W, DD, DD, m0, n0, DD, smem, acc);

    const int lane = threadIdx.x & 31;
    const int lr = lane & 15, hi = lane >> 4;
#pragma unroll
    for (int t = 0; t < 4; ++t) {
        const int n = n0 + 16 * t + lr;
        const float bn = bias[n];
#pragma unroll
        for (int s2 = 0; s2 < 2; ++s2)
#pragma unroll
            for (int r = 0; r < 8; ++r) {
                const int m = m0 + 16 * s2 + r + 8 * hi;
                float v = acc[s2][t][r] + bn + xp[(size_t)m * DD + n];
                y[(size_t)m * DD + n] = f2bf(v);
            }
    }
}

// ---------------------------------------------------------------------------
// Kernel 5: to_out GEMM + bias -> f32 output [BS x OUT]
// ---------------------------------------------------------------------------
__global__ void __launch_bounds__(256)
toout_gemm_kernel(const unsigned short* __restrict__ yb,
                  const unsigned short* __restrict__ W,
                  const float* __restrict__ bias,
                  float* __restrict__ out) {
    __shared__ unsigned short smem[2 * LDS_TILE_ELEMS];
    const int w  = threadIdx.x >> 5;
    const int m0 = blockIdx.y * 256 + w * 32;
    const int n0 = blockIdx.x * 64;

    v8f acc[2][4];
#pragma unroll
    for (int s2 = 0; s2 < 2; ++s2)
#pragma unroll
        for (int t = 0; t < 4; ++t) acc[s2][t] = v8f_zero();

    gemm32x64_tdm(yb, DD, W, DD, OUTD, m0, n0, DD, smem, acc);

    const int lane = threadIdx.x & 31;
    const int lr = lane & 15, hi = lane >> 4;
#pragma unroll
    for (int t = 0; t < 4; ++t) {
        const int n = n0 + 16 * t + lr;
        const float bn = bias[n];
#pragma unroll
        for (int s2 = 0; s2 < 2; ++s2)
#pragma unroll
            for (int r = 0; r < 8; ++r) {
                const int m = m0 + 16 * s2 + r + 8 * hi;
                out[(size_t)m * OUTD + n] = acc[s2][t][r] + bn;
            }
    }
}

// ---------------------------------------------------------------------------
// Host launcher
// ---------------------------------------------------------------------------
extern "C" void kernel_launch(void* const* d_in, const int* in_sizes, int n_in,
                              void* d_out, int out_size, void* d_ws, size_t ws_size,
                              hipStream_t stream) {
    const float* x        = (const float*)d_in[0];
    const float* mask     = (const float*)d_in[1];
    const float* pos_emb  = (const float*)d_in[2];
    const float* ln_gamma = (const float*)d_in[3];
    const float* ln_beta  = (const float*)d_in[4];
    const float* w_qkv    = (const float*)d_in[5];
    const float* b_qkv    = (const float*)d_in[6];
    const float* w_out    = (const float*)d_in[7];
    const float* b_out    = (const float*)d_in[8];
    const float* w_to     = (const float*)d_in[9];
    const float* b_to     = (const float*)d_in[10];
    float* out = (float*)d_out;

    size_t off = 0;
    auto carve = [&](size_t bytes) -> void* {
        void* p = (char*)d_ws + off;
        off += (bytes + 255) & ~(size_t)255;
        return p;
    };
    float*          xp_f32 = (float*)         carve((size_t)BS * DD * 4);
    unsigned short* xp_bf  = (unsigned short*)carve((size_t)BS * DD * 2);
    unsigned short* wqkv   = (unsigned short*)carve((size_t)3 * DD * DD * 2);
    unsigned short* wout   = (unsigned short*)carve((size_t)DD * DD * 2);
    unsigned short* wto    = (unsigned short*)carve((size_t)OUTD * DD * 2);
    unsigned short* qbuf   = (unsigned short*)carve((size_t)BS * DD * 2);
    unsigned short* kbuf   = (unsigned short*)carve((size_t)BS * DD * 2);
    unsigned short* vTbuf  = (unsigned short*)carve((size_t)BS * DD * 2);
    unsigned short* obuf   = (unsigned short*)carve((size_t)BS * DD * 2);
    unsigned short* ybuf   = (unsigned short*)carve((size_t)BS * DD * 2);
    (void)ws_size; (void)in_sizes; (void)n_in; (void)out_size;

    {
        int n1 = 3 * DD * DD, n2 = DD * DD, n3 = OUTD * DD;
        cvt_bf16_kernel<<<(n1 + 255) / 256, 256, 0, stream>>>(w_qkv, wqkv, n1);
        cvt_bf16_kernel<<<(n2 + 255) / 256, 256, 0, stream>>>(w_out, wout, n2);
        cvt_bf16_kernel<<<(n3 + 255) / 256, 256, 0, stream>>>(w_to,  wto,  n3);
    }
    ln_pos_kernel<<<BS, 256, 0, stream>>>(x, pos_emb, ln_gamma, ln_beta, xp_f32, xp_bf);
    qkv_gemm_kernel<<<dim3(3 * DD / 64, BS / 256), 256, 0, stream>>>(
        xp_bf, wqkv, b_qkv, qbuf, kbuf, vTbuf);
    attn_kernel<<<(BB * HH * (SS / 16)) / 8, 256, 0, stream>>>(
        qbuf, kbuf, vTbuf, mask, obuf);
    outproj_gemm_kernel<<<dim3(DD / 64, BS / 256), 256, 0, stream>>>(
        obuf, wout, b_out, xp_f32, ybuf);
    toout_gemm_kernel<<<dim3(OUTD / 64, BS / 256), 256, 0, stream>>>(
        ybuf, wto, b_to, out);
}
